// PagedMixtralBlock_88381837017443
// MI455X (gfx1250) — compile-verified, matrix operand order checked by reading, not run on hardware
//
#include <hip/hip_runtime.h>
#include <hip/hip_bf16.h>
#include <math.h>

// ---------------- problem constants ----------------
#define BB   2
#define SS   1024
#define DD   2048
#define HH   16
#define KVH_ 4
#define HD_  128
#define EE   8
#define FF_  7168
#define MTOK (BB*SS)
#define EPS_ 1e-5f
#define SPAD (SS + 64)          // padded key stride for transposed V cache

typedef __attribute__((ext_vector_type(16))) __bf16 v16bf;
typedef __attribute__((ext_vector_type(8)))  __bf16 v8bf;
typedef __attribute__((ext_vector_type(8)))  float  v8f;

// ---------------- small helpers ----------------
__device__ __forceinline__ unsigned short f2bf(float f) {
  unsigned u = __builtin_bit_cast(unsigned, f);
  u += 0x7FFFu + ((u >> 16) & 1u);            // round-to-nearest-even
  return (unsigned short)(u >> 16);
}
__device__ __forceinline__ float bf2f(unsigned short h) {
  unsigned u = ((unsigned)h) << 16;
  return __builtin_bit_cast(float, u);
}
__device__ __forceinline__ v8f wmma_bf16(v16bf a, v16bf b, v8f c) {
  return __builtin_amdgcn_wmma_f32_16x16x32_bf16(false, a, false, b,
                                                 (short)0, c, false, false);
}
// Build a 16-element bf16 fragment from two contiguous 16B chunks:
// elements [0..7] at base+8*half, [8..15] at base+16+8*half (CDNA5 ISA 7.12.2
// K-pair pattern for 16-bit A/B operands). Both chunks 16B aligned.
__device__ __forceinline__ v16bf ld_frag(const unsigned short* base, int half) {
  v8bf lo = __builtin_bit_cast(v8bf, *(const uint4*)(base + 8*half));
  v8bf hi = __builtin_bit_cast(v8bf, *(const uint4*)(base + 16 + 8*half));
  return __builtin_shufflevector(lo, hi, 0,1,2,3,4,5,6,7,8,9,10,11,12,13,14,15);
}

// ---- pipelined staging primitives (128-thread block, 64x32 A tile, 32x64 B tile) ----
// A: thread (r=tid>>1, part=tid&1) moves 32B of one row.
__device__ __forceinline__ void ldg_a(uint4& u0, uint4& u1,
                                      const unsigned short* src, bool valid) {
  if (valid) {
    const uint4* s = (const uint4*)src;
    u0 = s[0]; u1 = s[1];
  } else {
    u0 = make_uint4(0,0,0,0); u1 = u0;
  }
}
__device__ __forceinline__ void sts_a(unsigned short* As, uint4 u0, uint4 u1, int tid) {
  int r = tid >> 1, part = tid & 1;
  uint4* d = (uint4*)(As + r*40 + part*16);
  d[0] = u0; d[1] = u1;
}
// B: thread (n=tid&63, kq=tid>>6) reads a 16-deep K column (immediate offsets,
// N is compile-time), converts later, stores two b128 into N-major tile.
template <int N>
__device__ __forceinline__ void ldg_b(float (&f)[16], const float* __restrict__ Bw,
                                      int n0, int k0, int tid) {
  const float* col = Bw + (size_t)(k0 + (tid >> 6) * 16) * N + n0 + (tid & 63);
#pragma unroll
  for (int i = 0; i < 16; ++i) f[i] = col[(size_t)i * N];
}
__device__ __forceinline__ void sts_b(unsigned short* BsT, const float (&f)[16], int tid) {
  int n = tid & 63, kq = tid >> 6;
#pragma unroll
  for (int h = 0; h < 2; ++h) {
    unsigned pk[4];
#pragma unroll
    for (int i = 0; i < 4; ++i)
      pk[i] = (unsigned)f2bf(f[h*8 + 2*i]) | ((unsigned)f2bf(f[h*8 + 2*i + 1]) << 16);
    *(uint4*)(BsT + n*40 + kq*16 + h*8) = make_uint4(pk[0], pk[1], pk[2], pk[3]);
  }
}

// ---------------- RMSNorm -> bf16 ----------------
__global__ void k_rmsnorm(const float* __restrict__ x, const float* __restrict__ w,
                          unsigned short* __restrict__ out) {
  int row = blockIdx.x;
  const float* xr = x + (size_t)row * DD;
  __shared__ float red[256];
  float s = 0.f;
  for (int i = threadIdx.x; i < DD; i += 256) { float v = xr[i]; s += v * v; }
  red[threadIdx.x] = s; __syncthreads();
  for (int off = 128; off > 0; off >>= 1) {
    if (threadIdx.x < (unsigned)off) red[threadIdx.x] += red[threadIdx.x + off];
    __syncthreads();
  }
  float inv = rsqrtf(red[0] / (float)DD + EPS_);
  for (int i = threadIdx.x; i < DD; i += 256)
    out[(size_t)row * DD + i] = f2bf(xr[i] * inv * w[i]);
}

// ------------- dense WMMA GEMM (M=MTOK): C = A(bf16) @ W(f32->bf16) [+resid] -------------
template <int N, int Kd, bool RESID>
__global__ void __launch_bounds__(128)
k_gemm(const unsigned short* __restrict__ A, const float* __restrict__ Bw,
       float* __restrict__ C, const float* __restrict__ resid) {
  __shared__ __align__(16) unsigned short As[2][64*40];
  __shared__ __align__(16) unsigned short BsT[2][64*40];
  int n0 = blockIdx.x * 64, m0 = blockIdx.y * 64;
  int tid = threadIdx.x, lane = tid & 31, wave = tid >> 5;
  int wm = (wave & 1) * 32, wn = (wave >> 1) * 32;
  int nl = lane & 15, half = lane >> 4;
  const unsigned short* arow = A + (size_t)(m0 + (tid >> 1)) * Kd + (tid & 1) * 16;
  v8f acc[2][2];
#pragma unroll
  for (int i = 0; i < 2; ++i)
#pragma unroll
    for (int j = 0; j < 2; ++j) acc[i][j] = (v8f){0,0,0,0,0,0,0,0};

  uint4 a0, a1; float fb[16];
  ldg_a(a0, a1, arow, true);
  ldg_b<N>(fb, Bw, n0, 0, tid);
  sts_a(As[0], a0, a1, tid);
  sts_b(BsT[0], fb, tid);
  __syncthreads();
  int p = 0;
  for (int k0 = 0; k0 < Kd; k0 += 32) {
    bool more = (k0 + 32) < Kd;
    if (more) {                              // issue next-tile global loads early
      ldg_a(a0, a1, arow + k0 + 32, true);
      ldg_b<N>(fb, Bw, n0, k0 + 32, tid);
    }
    v16bf af0 = ld_frag(As[p]  + (wm + 0  + nl)*40, half);
    v16bf af1 = ld_frag(As[p]  + (wm + 16 + nl)*40, half);
    v16bf bf0 = ld_frag(BsT[p] + (wn + 0  + nl)*40, half);
    v16bf bf1 = ld_frag(BsT[p] + (wn + 16 + nl)*40, half);
    acc[0][0] = wmma_bf16(af0, bf0, acc[0][0]);
    acc[0][1] = wmma_bf16(af0, bf1, acc[0][1]);
    acc[1][0] = wmma_bf16(af1, bf0, acc[1][0]);
    acc[1][1] = wmma_bf16(af1, bf1, acc[1][1]);
    if (more) {                              // convert + store after the MMAs
      sts_a(As[p^1], a0, a1, tid);
      sts_b(BsT[p^1], fb, tid);
    }
    __syncthreads();
    p ^= 1;
  }
#pragma unroll
  for (int i = 0; i < 2; ++i)
#pragma unroll
    for (int j = 0; j < 2; ++j)
#pragma unroll
      for (int r = 0; r < 8; ++r) {
        int gm = m0 + wm + i*16 + r + 8*half;
        int gn = n0 + wn + j*16 + nl;
        float v = acc[i][j][r];
        if (RESID) v += resid[(size_t)gm * N + gn];
        C[(size_t)gm * N + gn] = v;
      }
}

// ---------------- RoPE + transpose to [b][h][S][HD], f32 -> bf16 ----------------
__global__ void k_rope(const float* __restrict__ src, unsigned short* __restrict__ dst,
                       int nh) {
  int idx = blockIdx.x * 256 + threadIdx.x;     // pair index
  int total = MTOK * nh * (HD_/2);
  if (idx >= total) return;
  int p    = idx % (HD_/2);
  int head = (idx / (HD_/2)) % nh;
  int tok  = idx / ((HD_/2) * nh);
  int b = tok / SS, s = tok % SS;
  float inv = __powf(1.0e6f, -(float)(2*p) / (float)HD_);
  float ang = (float)s * inv;
  float sn, cs; __sincosf(ang, &sn, &cs);
  const float* pr = src + (size_t)tok * nh * HD_ + head * HD_ + 2*p;
  float x1 = pr[0], x2 = pr[1];
  size_t o = (((size_t)(b*nh + head)) * SS + s) * HD_ + 2*p;
  dst[o]   = f2bf(x1*cs - x2*sn);
  dst[o+1] = f2bf(x1*sn + x2*cs);
}

// ------- V transpose/convert to [b][kvh][HD][SPAD] bf16, zero-padded tail -------
__global__ void k_cvt_v(const float* __restrict__ v32, unsigned short* __restrict__ vb) {
  int idx = blockIdx.x * 256 + threadIdx.x;
  int total = BB * KVH_ * HD_ * SPAD;
  if (idx >= total) return;
  int s    = idx % SPAD;
  int hd   = (idx / SPAD) % HD_;
  int head = (idx / (SPAD * HD_)) % KVH_;
  int b    = idx / (SPAD * HD_ * KVH_);
  unsigned short val = 0;
  if (s < SS)
    val = f2bf(v32[(size_t)(b*SS + s) * KVH_ * HD_ + head * HD_ + hd]);
  vb[idx] = val;
}

// ---------------- single-wave flash attention, 16 q-rows, 32 keys/iter ----------------
__global__ void __launch_bounds__(32)
k_attn(const unsigned short* __restrict__ Q,   // [B][H][S][HD]
       const unsigned short* __restrict__ Kc,  // [B][KVH][S][HD]
       const unsigned short* __restrict__ Vc,  // [B][KVH][HD][SPAD]
       unsigned short* __restrict__ O) {       // [tok][H*HD]
  int q0 = blockIdx.x * 16;
  int h  = blockIdx.y, b = blockIdx.z;
  int hkv = h / (HH / KVH_);
  int lane = threadIdx.x, nl = lane & 15, half = lane >> 4;
  __shared__ __align__(16) unsigned short Ps[16*40];
  const unsigned short* qbase = Q  + (((size_t)(b*HH   + h  )) * SS + q0) * HD_;
  const unsigned short* kbase = Kc + (((size_t)(b*KVH_ + hkv)) * SS) * HD_;
  const unsigned short* vbase = Vc + ((size_t)(b*KVH_ + hkv)) * HD_ * SPAD;

  v16bf aq[4];
#pragma unroll
  for (int c = 0; c < 4; ++c)
    aq[c] = ld_frag(qbase + (size_t)nl * HD_ + c*32, half);

  v8f acc[8];
#pragma unroll
  for (int t = 0; t < 8; ++t) acc[t] = (v8f){0,0,0,0,0,0,0,0};
  float rm[8], rl[8];
#pragma unroll
  for (int r = 0; r < 8; ++r) { rm[r] = -INFINITY; rl[r] = 0.f; }

  const float scale = 0.08838834764831845f;    // 1/sqrt(128)
  int ktmax = (q0 + 15) >> 5;
  for (int kt = 0; kt <= ktmax; ++kt) {
    int kb0 = kt * 32;
    int key0 = kb0 + nl;      if (key0 > SS-1) key0 = SS-1;
    int key1 = kb0 + 16 + nl; if (key1 > SS-1) key1 = SS-1;
    v8f s0 = (v8f){0,0,0,0,0,0,0,0}, s1 = (v8f){0,0,0,0,0,0,0,0};
#pragma unroll
    for (int c = 0; c < 4; ++c) {
      v16bf bk0 = ld_frag(kbase + (size_t)key0 * HD_ + c*32, half);
      v16bf bk1 = ld_frag(kbase + (size_t)key1 * HD_ + c*32, half);
      s0 = wmma_bf16(aq[c], bk0, s0);
      s1 = wmma_bf16(aq[c], bk1, s1);
    }
#pragma unroll
    for (int r = 0; r < 8; ++r) {
      int qrow = r + 8*half, qpos = q0 + qrow;
      float v0 = (kb0 + nl      <= qpos) ? s0[r]*scale : -INFINITY;
      float v1 = (kb0 + 16 + nl <= qpos) ? s1[r]*scale : -INFINITY;
      float mx = fmaxf(v0, v1);
#pragma unroll
      for (int off = 1; off < 16; off <<= 1) mx = fmaxf(mx, __shfl_xor(mx, off));
      float mnew = fmaxf(rm[r], mx);
      float corr = __expf(rm[r] - mnew);
      float p0 = __expf(v0 - mnew), p1 = __expf(v1 - mnew);
      float ps = p0 + p1;
#pragma unroll
      for (int off = 1; off < 16; off <<= 1) ps += __shfl_xor(ps, off);
      rl[r] = rl[r]*corr + ps;
      rm[r] = mnew;
#pragma unroll
      for (int t = 0; t < 8; ++t) acc[t][r] *= corr;
      Ps[qrow*40 + nl]      = f2bf(p0);
      Ps[qrow*40 + 16 + nl] = f2bf(p1);
    }
    __syncthreads();
    v16bf ap = ld_frag(Ps + nl*40, half);
    __syncthreads();
#pragma unroll
    for (int t = 0; t < 8; ++t) {
      // V fragment: contiguous keys from transposed, zero-padded cache.
      v16bf bv = ld_frag(vbase + (size_t)(t*16 + nl) * SPAD + kb0, half);
      acc[t] = wmma_bf16(ap, bv, acc[t]);
    }
  }
#pragma unroll
  for (int t = 0; t < 8; ++t)
#pragma unroll
    for (int r = 0; r < 8; ++r) {
      int qrow = r + 8*half;
      size_t tok = (size_t)b * SS + q0 + qrow;
      O[tok * (HH*HD_) + h*HD_ + t*16 + nl] = f2bf(acc[t][r] / rl[r]);
    }
}

// ---------------- gate: logits, top-2 softmax, routing lists ----------------
__global__ void k_gate(const unsigned short* __restrict__ h2, const float* __restrict__ gw,
                       int* __restrict__ cnt, int* __restrict__ tok_list,
                       float* __restrict__ gate_list) {
  int row = blockIdx.x;
  __shared__ float red[128];
  __shared__ float logits[EE];
  float acc[EE];
#pragma unroll
  for (int e = 0; e < EE; ++e) acc[e] = 0.f;
  for (int d = threadIdx.x; d < DD; d += 128) {
    float hv = bf2f(h2[(size_t)row * DD + d]);
#pragma unroll
    for (int e = 0; e < EE; ++e) acc[e] += hv * gw[d*EE + e];
  }
  for (int e = 0; e < EE; ++e) {
    red[threadIdx.x] = acc[e]; __syncthreads();
    for (int off = 64; off > 0; off >>= 1) {
      if (threadIdx.x < (unsigned)off) red[threadIdx.x] += red[threadIdx.x + off];
      __syncthreads();
    }
    if (threadIdx.x == 0) logits[e] = red[0];
    __syncthreads();
  }
  if (threadIdx.x == 0) {
    int i0 = 0;
    for (int e = 1; e < EE; ++e) if (logits[e] > logits[i0]) i0 = e;
    int i1 = (i0 == 0) ? 1 : 0;
    for (int e = 0; e < EE; ++e)
      if (e != i0 && logits[e] > logits[i1]) i1 = e;
    float m  = fmaxf(logits[i0], logits[i1]);
    float g0 = __expf(logits[i0] - m), g1 = __expf(logits[i1] - m);
    float inv = 1.f / (g0 + g1);
    g0 *= inv; g1 *= inv;
    int s0 = atomicAdd(&cnt[i0], 1);
    tok_list[i0*MTOK + s0] = row; gate_list[i0*MTOK + s0] = g0;
    int s1 = atomicAdd(&cnt[i1], 1);
    tok_list[i1*MTOK + s1] = row; gate_list[i1*MTOK + s1] = g1;
  }
}

__global__ void k_zero_cnt(int* cnt) { if (threadIdx.x < EE) cnt[threadIdx.x] = 0; }

// ---------------- MoE up: t = silu(h@w1) * (h@w3), gathered rows ----------------
__global__ void __launch_bounds__(128)
k_moe_up(const unsigned short* __restrict__ h2, const float* __restrict__ w1,
         const float* __restrict__ w3, const int* __restrict__ cnt,
         const int* __restrict__ tok_list, unsigned short* __restrict__ T) {
  int e = blockIdx.z;
  int rows = cnt[e];
  int m0 = blockIdx.y * 64;
  if (m0 >= rows) return;
  int n0 = blockIdx.x * 64;
  const float* B1 = w1 + (size_t)e * DD * FF_;
  const float* B3 = w3 + (size_t)e * DD * FF_;
  const int*   tl = tok_list + e * MTOK;
  __shared__ __align__(16) unsigned short As[2][64*40];
  __shared__ __align__(16) unsigned short B1s[2][64*40];
  __shared__ __align__(16) unsigned short B3s[2][64*40];
  __shared__ int rowtok[64];
  int tid = threadIdx.x, lane = tid & 31, wave = tid >> 5;
  int wm = (wave & 1) * 32, wn = (wave >> 1) * 32;
  int nl = lane & 15, half = lane >> 4;
  if (tid < 64) rowtok[tid] = (m0 + tid < rows) ? tl[m0 + tid] : -1;
  __syncthreads();
  int mytok = rowtok[tid >> 1];
  bool rowOk = mytok >= 0;
  const unsigned short* arow = h2 + (size_t)(rowOk ? mytok : 0) * DD + (tid & 1) * 16;
  v8f a1[2][2], a3[2][2];
#pragma unroll
  for (int i = 0; i < 2; ++i)
#pragma unroll
    for (int j = 0; j < 2; ++j) {
      a1[i][j] = (v8f){0,0,0,0,0,0,0,0};
      a3[i][j] = (v8f){0,0,0,0,0,0,0,0};
    }
  uint4 u0, u1; float f1[16], f3[16];
  ldg_a(u0, u1, arow, rowOk);
  ldg_b<FF_>(f1, B1, n0, 0, tid);
  ldg_b<FF_>(f3, B3, n0, 0, tid);
  sts_a(As[0], u0, u1, tid);
  sts_b(B1s[0], f1, tid);
  sts_b(B3s[0], f3, tid);
  __syncthreads();
  int p = 0;
  for (int k0 = 0; k0 < DD; k0 += 32) {
    bool more = (k0 + 32) < DD;
    if (more) {
      ldg_a(u0, u1, arow + k0 + 32, rowOk);
      ldg_b<FF_>(f1, B1, n0, k0 + 32, tid);
      ldg_b<FF_>(f3, B3, n0, k0 + 32, tid);
    }
    v16bf af0 = ld_frag(As[p]  + (wm + 0  + nl)*40, half);
    v16bf af1 = ld_frag(As[p]  + (wm + 16 + nl)*40, half);
    v16bf b10 = ld_frag(B1s[p] + (wn + 0  + nl)*40, half);
    v16bf b11 = ld_frag(B1s[p] + (wn + 16 + nl)*40, half);
    v16bf b30 = ld_frag(B3s[p] + (wn + 0  + nl)*40, half);
    v16bf b31 = ld_frag(B3s[p] + (wn + 16 + nl)*40, half);
    a1[0][0] = wmma_bf16(af0, b10, a1[0][0]);
    a1[0][1] = wmma_bf16(af0, b11, a1[0][1]);
    a1[1][0] = wmma_bf16(af1, b10, a1[1][0]);
    a1[1][1] = wmma_bf16(af1, b11, a1[1][1]);
    a3[0][0] = wmma_bf16(af0, b30, a3[0][0]);
    a3[0][1] = wmma_bf16(af0, b31, a3[0][1]);
    a3[1][0] = wmma_bf16(af1, b30, a3[1][0]);
    a3[1][1] = wmma_bf16(af1, b31, a3[1][1]);
    if (more) {
      sts_a(As[p^1], u0, u1, tid);
      sts_b(B1s[p^1], f1, tid);
      sts_b(B3s[p^1], f3, tid);
    }
    __syncthreads();
    p ^= 1;
  }
#pragma unroll
  for (int i = 0; i < 2; ++i)
#pragma unroll
    for (int j = 0; j < 2; ++j)
#pragma unroll
      for (int r = 0; r < 8; ++r) {
        int lrow = wm + i*16 + r + 8*half;
        if (m0 + lrow < rows) {
          float g = a1[i][j][r];
          float tv = g * (1.f / (1.f + __expf(-g))) * a3[i][j][r];
          T[((size_t)e * MTOK + m0 + lrow) * FF_ + n0 + wn + j*16 + nl] = f2bf(tv);
        }
      }
}

// ---------------- MoE down: out += gate * (t @ w2) ----------------
__global__ void __launch_bounds__(128)
k_moe_down(const unsigned short* __restrict__ T, const float* __restrict__ w2,
           const int* __restrict__ cnt, const int* __restrict__ tok_list,
           const float* __restrict__ gate_list, float* __restrict__ out) {
  int e = blockIdx.z;
  int rows = cnt[e];
  int m0 = blockIdx.y * 64;
  if (m0 >= rows) return;
  int n0 = blockIdx.x * 64;
  const float* Bw = w2 + (size_t)e * FF_ * DD;
  const int*   tl = tok_list + e * MTOK;
  const float* gl = gate_list + e * MTOK;
  const unsigned short* A = T + (size_t)e * MTOK * FF_;
  __shared__ __align__(16) unsigned short As[2][64*40];
  __shared__ __align__(16) unsigned short BsT[2][64*40];
  int tid = threadIdx.x, lane = tid & 31, wave = tid >> 5;
  int wm = (wave & 1) * 32, wn = (wave >> 1) * 32;
  int nl = lane & 15, half = lane >> 4;
  bool rowOk = (m0 + (tid >> 1)) < rows;
  const unsigned short* arow = A + (size_t)(m0 + (tid >> 1)) * FF_ + (tid & 1) * 16;
  v8f acc[2][2];
#pragma unroll
  for (int i = 0; i < 2; ++i)
#pragma unroll
    for (int j = 0; j < 2; ++j) acc[i][j] = (v8f){0,0,0,0,0,0,0,0};
  uint4 u0, u1; float fb[16];
  ldg_a(u0, u1, arow, rowOk);
  ldg_b<DD>(fb, Bw, n0, 0, tid);
  sts_a(As[0], u0, u1, tid);
  sts_b(BsT[0], fb, tid);
  __syncthreads();
  int p = 0;
  for (int k0 = 0; k0 < FF_; k0 += 32) {
    bool more = (k0 + 32) < FF_;
    if (more) {
      ldg_a(u0, u1, arow + k0 + 32, rowOk);
      ldg_b<DD>(fb, Bw, n0, k0 + 32, tid);
    }
    v16bf af0 = ld_frag(As[p]  + (wm + 0  + nl)*40, half);
    v16bf af1 = ld_frag(As[p]  + (wm + 16 + nl)*40, half);
    v16bf bf0 = ld_frag(BsT[p] + (wn + 0  + nl)*40, half);
    v16bf bf1 = ld_frag(BsT[p] + (wn + 16 + nl)*40, half);
    acc[0][0] = wmma_bf16(af0, bf0, acc[0][0]);
    acc[0][1] = wmma_bf16(af0, bf1, acc[0][1]);
    acc[1][0] = wmma_bf16(af1, bf0, acc[1][0]);
    acc[1][1] = wmma_bf16(af1, bf1, acc[1][1]);
    if (more) {
      sts_a(As[p^1], u0, u1, tid);
      sts_b(BsT[p^1], fb, tid);
    }
    __syncthreads();
    p ^= 1;
  }
#pragma unroll
  for (int i = 0; i < 2; ++i)
#pragma unroll
    for (int j = 0; j < 2; ++j)
#pragma unroll
      for (int r = 0; r < 8; ++r) {
        int lrow = wm + i*16 + r + 8*half;
        int slot = m0 + lrow;
        if (slot < rows) {
          int tok = tl[slot];
          float g = gl[slot];
          atomicAdd(&out[(size_t)tok * DD + n0 + wn + j*16 + nl], g * acc[i][j][r]);
        }
      }
}

// ---------------- out = x2 (residual init) ----------------
__global__ void k_copy(const float* __restrict__ src, float* __restrict__ dst, int n) {
  int i = blockIdx.x * 256 + threadIdx.x;
  if (i < n) dst[i] = src[i];
}

// ---------------- host launcher ----------------
extern "C" void kernel_launch(void* const* d_in, const int* in_sizes, int n_in,
                              void* d_out, int out_size, void* d_ws, size_t ws_size,
                              hipStream_t stream) {
  (void)in_sizes; (void)n_in; (void)out_size; (void)ws_size;
  const float* x       = (const float*)d_in[0];
  const float* ln_w    = (const float*)d_in[1];
  const float* ff_ln_w = (const float*)d_in[2];
  const float* wq      = (const float*)d_in[3];
  const float* wk      = (const float*)d_in[4];
  const float* wv      = (const float*)d_in[5];
  const float* wo      = (const float*)d_in[6];
  const float* gate_w  = (const float*)d_in[7];
  const float* w1      = (const float*)d_in[8];
  const float* w3      = (const float*)d_in[9];
  const float* w2      = (const float*)d_in[10];
  float* out = (float*)d_out;

  char* ws = (char*)d_ws;
  size_t off = 0;
  auto alloc = [&](size_t bytes) -> void* {
    void* p = ws + off;
    off = (off + bytes + 255) & ~(size_t)255;
    return p;
  };
  unsigned short* hb   = (unsigned short*)alloc((size_t)MTOK * DD * 2);
  float*          q32  = (float*)alloc((size_t)MTOK * HH * HD_ * 4);
  float*          k32  = (float*)alloc((size_t)MTOK * KVH_ * HD_ * 4);
  float*          v32  = (float*)alloc((size_t)MTOK * KVH_ * HD_ * 4);
  unsigned short* qb   = (unsigned short*)alloc((size_t)MTOK * HH * HD_ * 2);
  unsigned short* kb   = (unsigned short*)alloc((size_t)MTOK * KVH_ * HD_ * 2);
  unsigned short* vb   = (unsigned short*)alloc((size_t)BB * KVH_ * HD_ * SPAD * 2);
  unsigned short* ob   = (unsigned short*)alloc((size_t)MTOK * HH * HD_ * 2);
  float*          x2   = (float*)alloc((size_t)MTOK * DD * 4);
  unsigned short* h2b  = (unsigned short*)alloc((size_t)MTOK * DD * 2);
  int*            cnt  = (int*)alloc(EE * 4);
  int*            tokl = (int*)alloc((size_t)EE * MTOK * 4);
  float*          gl   = (float*)alloc((size_t)EE * MTOK * 4);
  unsigned short* T    = (unsigned short*)alloc((size_t)EE * MTOK * FF_ * 2);

  // 1. pre-attention RMSNorm (fp32 x -> bf16 h)
  k_rmsnorm<<<MTOK, 256, 0, stream>>>(x, ln_w, hb);
  // 2. q/k/v projections (WMMA bf16, weights converted during LDS staging)
  k_gemm<HH*HD_,   DD, false><<<dim3((HH*HD_)/64,   MTOK/64), 128, 0, stream>>>(hb, wq, q32, nullptr);
  k_gemm<KVH_*HD_, DD, false><<<dim3((KVH_*HD_)/64, MTOK/64), 128, 0, stream>>>(hb, wk, k32, nullptr);
  k_gemm<KVH_*HD_, DD, false><<<dim3((KVH_*HD_)/64, MTOK/64), 128, 0, stream>>>(hb, wv, v32, nullptr);
  // 3. RoPE + head-major bf16 layouts; V transposed+padded
  k_rope<<<(MTOK*HH  *(HD_/2) + 255)/256, 256, 0, stream>>>(q32, qb, HH);
  k_rope<<<(MTOK*KVH_*(HD_/2) + 255)/256, 256, 0, stream>>>(k32, kb, KVH_);
  k_cvt_v<<<(BB*KVH_*HD_*SPAD + 255)/256, 256, 0, stream>>>(v32, vb);
  // 4. causal flash attention (GQA 4:1)
  k_attn<<<dim3(SS/16, HH, BB), 32, 0, stream>>>(qb, kb, vb, ob);
  // 5. output projection + residual
  k_gemm<DD, HH*HD_, true><<<dim3(DD/64, MTOK/64), 128, 0, stream>>>(ob, wo, x2, x);
  // 6. pre-MoE RMSNorm
  k_rmsnorm<<<MTOK, 256, 0, stream>>>(x2, ff_ln_w, h2b);
  // 7. routing
  k_zero_cnt<<<1, 32, 0, stream>>>(cnt);
  k_gate<<<MTOK, 128, 0, stream>>>(h2b, gate_w, cnt, tokl, gl);
  // 8. expert up-projection (w1,w3 fused, SiLU*mul)
  k_moe_up<<<dim3(FF_/64, MTOK/64, EE), 128, 0, stream>>>(h2b, w1, w3, cnt, tokl, T);
  // 9. residual init of output, then expert down-projection accumulate
  k_copy<<<(MTOK*DD + 255)/256, 256, 0, stream>>>(x2, out, MTOK*DD);
  k_moe_down<<<dim3(DD/64, MTOK/64, EE), 128, 0, stream>>>(T, w2, cnt, tokl, gl, out);
}
